// ExLoss_28870770164354
// MI455X (gfx1250) — matrix operand order repaired and checked by reading, not compiled, not run on hardware
//
#include <hip/hip_runtime.h>
#include <math.h>

typedef __attribute__((ext_vector_type(2))) float v2f;
typedef __attribute__((ext_vector_type(8))) float v8f;
typedef __bf16 bf16_t;
typedef __attribute__((ext_vector_type(8))) __bf16 v8bf;
typedef __attribute__((ext_vector_type(16))) __bf16 v16bf;
typedef unsigned int u32;
typedef __attribute__((ext_vector_type(4))) u32 u32x4;
typedef __attribute__((ext_vector_type(8))) int i32x8;
typedef __attribute__((ext_vector_type(4))) int i32x4;

#define B_SZ 256
#define D_SZ 2048
#define C_SZ 16384
#define MARGINF 0.3f
#define BIGV 1e9f

__device__ __forceinline__ float softplus_f(float x) {
  if (x > 20.f) return x;
  if (x < -20.f) return expf(x);
  return log1pf(expf(x));
}

__device__ __forceinline__ void split_bf16(float v, bf16_t& h, bf16_t& l) {
  h = (bf16_t)v;
  l = (bf16_t)(v - (float)h);
}

// --- TDM: 2-D tile load Global -> LDS (bf16 elements, data_size=2B) ----------
// D# per CDNA5 ISA: group0 = {count|..., lds_addr,
// global_addr[31:0], global_addr[56:32]|type=2}; group1 packs data_size,
// tensor_dim0/1, tile_dim0/1, tensor_dim0_stride. Groups 2/3 zero (2-D tile).
// This toolchain's builtin is the 6-arg clang-23 form:
//   (u32x4 g0, i32x8 g1, i32x4, i32x4, i32x8, i32 cpol)
__device__ __forceinline__ void tdm_load_2d_bf16(u32 lds_off, const bf16_t* gptr,
                                                 u32 tensor_d0, u32 tensor_d1,
                                                 u32 tile_d0, u32 tile_d1,
                                                 u32 stride0) {
  unsigned long long ga = (unsigned long long)(uintptr_t)gptr;
  u32x4 g0;
  g0[0] = 1u;                                  // count=1, user mode
  g0[1] = lds_off;                             // lds_addr (bytes)
  g0[2] = (u32)ga;                             // global_addr[31:0]
  g0[3] = (u32)(ga >> 32) | (2u << 30);        // global_addr[56:32] | type=2
  i32x8 g1;
  g1[0] = (int)(1u << 16);                     // workgroup_mask=0, data_size=1 (2B)
  g1[1] = (int)((tensor_d0 & 0xFFFFu) << 16);  // atomic_barrier=0, dim0[15:0]
  g1[2] = (int)((tensor_d0 >> 16) | ((tensor_d1 & 0xFFFFu) << 16));
  g1[3] = (int)((tensor_d1 >> 16) | (tile_d0 << 16));
  g1[4] = (int)(tile_d1 & 0xFFFFu);            // tile_dim1; tile_dim2=0
  g1[5] = (int)stride0;                        // tensor_dim0_stride[31:0]
  g1[6] = 0;
  g1[7] = 0;
  i32x4 z4 = {0, 0, 0, 0};
  i32x8 z8 = {0, 0, 0, 0, 0, 0, 0, 0};
  __builtin_amdgcn_tensor_load_to_lds(g0, g1, z4, z4, z8, 0);
}

// WMMA bf16 fragment loaders (16-bit A 16x32 / B 32x16 layouts, wave32)
__device__ __forceinline__ v16bf fragA(const char* smem, int off, int row, int hi) {
  const bf16_t* base = (const bf16_t*)(smem + off);
  v8bf p0 = *(const v8bf*)&base[row * 32 + 8 * hi];        // K = 8*hi .. +7
  v8bf p1 = *(const v8bf*)&base[row * 32 + 16 + 8 * hi];   // K = 16+8*hi .. +7
  return __builtin_shufflevector(p0, p1, 0, 1, 2, 3, 4, 5, 6, 7, 8, 9, 10, 11,
                                 12, 13, 14, 15);
}
__device__ __forceinline__ v16bf fragB(const char* smem, int off, int n, int hi) {
  const bf16_t* base = (const bf16_t*)(smem + off);
  v8bf p0 = *(const v8bf*)&base[n * 32 + 8 * hi];          // K = 8*hi .. +7
  v8bf p1 = *(const v8bf*)&base[n * 32 + 16 + 8 * hi];     // K = 16+8*hi .. +7
  return __builtin_shufflevector(p0, p1, 0, 1, 2, 3, 4, 5, 6, 7, 8, 9, 10, 11,
                                 12, 13, 14, 15);
}

// ---------------------------------------------------------------------------
// Kernel 1: norms; X (f32, for sim), X_hi/X_lo (bf16 split, for big GEMM);
//           zero accumulators
// ---------------------------------------------------------------------------
__global__ __launch_bounds__(256) void prep_k(const float* __restrict__ inputs,
                                              float* __restrict__ Xf,
                                              bf16_t* __restrict__ Xh,
                                              bf16_t* __restrict__ Xl,
                                              float* __restrict__ acc) {
  __shared__ float red[256];
  const int r = blockIdx.x, t = threadIdx.x;
  if (r == 0 && t < 8) acc[t] = 0.f;
  float ss = 0.f;
  for (int k = t; k < D_SZ; k += 256) {
    float v = inputs[(size_t)r * D_SZ + k];
    ss += v * v;
  }
  red[t] = ss;
  __syncthreads();
  for (int off = 128; off > 0; off >>= 1) {
    if (t < off) red[t] += red[t + off];
    __syncthreads();
  }
  const float rinv = rsqrtf(red[0]);
  for (int k = t; k < D_SZ; k += 256) {
    float v = inputs[(size_t)r * D_SZ + k];
    float vn = v * rinv;
    size_t i0 = (size_t)r * D_SZ + k;
    size_t i1 = (size_t)(B_SZ + r) * D_SZ + k;
    Xf[i0] = v;
    Xf[i1] = vn;
    bf16_t h, l;
    split_bf16(v, h, l);
    Xh[i0] = h; Xl[i0] = l;
    split_bf16(vn, h, l);
    Xh[i1] = h; Xl[i1] = l;
  }
}

// ---------------------------------------------------------------------------
// Kernel 2: fused Y = V @ X.T (16384 x 512, K=2048), split-bf16 via
// V_WMMA_F32_16X16X32_BF16 (hi*hi + hi*lo + lo*hi, f32 accum).
// X tiles streamed by TENSOR_LOAD_TO_LDS (double-buffered, s_wait_tensorcnt);
// V tile f32->bf16 split on the fly (V read exactly once: 128 MB).
//   cols 0..255   -> outputs.T scatter-stored to d_out[1 + j*C + c]
//   cols 256..511 -> tsims kept in LDS, per-class mining -> acc[2]
// ---------------------------------------------------------------------------
#define KC 32
#define OFF_ASH 0
#define OFF_ASL 2048
#define OFF_XS 4096
#define XS_BUF_BYTES (512 * KC * 2)              // 32 KB per buffer
#define OFF_XSH(buf) (OFF_XS + (buf) * XS_BUF_BYTES)
#define OFF_XSL(buf) (OFF_XS + 2 * XS_BUF_BYTES + (buf) * XS_BUF_BYTES)
#define SMEM_BYTES (OFF_XS + 4 * XS_BUF_BYTES)   // 132 KB

__global__ __launch_bounds__(256) void vx_gemm_mine_k(const float* __restrict__ V,
                                                      const bf16_t* __restrict__ Xh,
                                                      const bf16_t* __restrict__ Xl,
                                                      const int* __restrict__ targets,
                                                      float* __restrict__ out,
                                                      float* __restrict__ acc) {
  extern __shared__ char smem[];
  float* tsbuf = (float*)(smem + OFF_XS);  // 32 KB, reused after GEMM

  const int t = threadIdx.x;
  const int lane = t & 31;
  const int w = t >> 5;
  const int c0 = blockIdx.x * 32;
  const int mt = w & 1;
  const int nb = (w >> 1) * 128;
  const int hi = lane >> 4;
  const int lo = lane & 15;

  v8f accv[8];
#pragma unroll
  for (int i = 0; i < 8; ++i) accv[i] = (v8f)0.f;

  const int NCHUNK = D_SZ / KC;
  // prologue: TDM loads for chunk 0 into buffer 0 (wave 0 only; per-wave op)
  if (t < 32) {
    tdm_load_2d_bf16(OFF_XSH(0), Xh, D_SZ, 512, KC, 512, D_SZ);
    tdm_load_2d_bf16(OFF_XSL(0), Xl, D_SZ, 512, KC, 512, D_SZ);
  }

  int buf = 0;
#pragma unroll 1
  for (int c = 0; c < NCHUNK; ++c) {
    const int k0 = c * KC;
    if (t < 32) __builtin_amdgcn_s_wait_tensorcnt(0);
    __syncthreads();  // X chunk c landed; all waves done with buf^1 and As
    if (t < 32 && c + 1 < NCHUNK) {  // prefetch chunk c+1 into other buffer
      tdm_load_2d_bf16(OFF_XSH(buf ^ 1), Xh + (k0 + KC), D_SZ, 512, KC, 512, D_SZ);
      tdm_load_2d_bf16(OFF_XSL(buf ^ 1), Xl + (k0 + KC), D_SZ, 512, KC, 512, D_SZ);
    }
    {  // stage V tile 32x32 f32 -> bf16 hi/lo (1 float4 per thread)
      const int r = t >> 3, kk = (t & 7) * 4;
      const float4 v4 = *(const float4*)&V[(size_t)(c0 + r) * D_SZ + k0 + kk];
      bf16_t* ash = (bf16_t*)(smem + OFF_ASH);
      bf16_t* asl = (bf16_t*)(smem + OFF_ASL);
      bf16_t h, l;
      split_bf16(v4.x, h, l); ash[r * KC + kk + 0] = h; asl[r * KC + kk + 0] = l;
      split_bf16(v4.y, h, l); ash[r * KC + kk + 1] = h; asl[r * KC + kk + 1] = l;
      split_bf16(v4.z, h, l); ash[r * KC + kk + 2] = h; asl[r * KC + kk + 2] = l;
      split_bf16(v4.w, h, l); ash[r * KC + kk + 3] = h; asl[r * KC + kk + 3] = l;
      if (c + 1 < NCHUNK)  // global_prefetch for next V chunk
        __builtin_prefetch(&V[(size_t)(c0 + r) * D_SZ + k0 + KC + kk], 0, 0);
    }
    __syncthreads();

    const v16bf ah = fragA(smem, OFF_ASH, mt * 16 + lo, hi);
    const v16bf al = fragA(smem, OFF_ASL, mt * 16 + lo, hi);
#pragma unroll
    for (int nt = 0; nt < 8; ++nt) {
      const int n = nb + nt * 16 + lo;
      const v16bf bh = fragB(smem, OFF_XSH(buf), n, hi);
      const v16bf bl = fragB(smem, OFF_XSL(buf), n, hi);
      accv[nt] = __builtin_amdgcn_wmma_f32_16x16x32_bf16(false, ah, false, bh,
                                                         (short)0, accv[nt],
                                                         false, false);
      accv[nt] = __builtin_amdgcn_wmma_f32_16x16x32_bf16(false, ah, false, bl,
                                                         (short)0, accv[nt],
                                                         false, false);
      accv[nt] = __builtin_amdgcn_wmma_f32_16x16x32_bf16(false, al, false, bh,
                                                         (short)0, accv[nt],
                                                         false, false);
    }
    buf ^= 1;
  }
  __syncthreads();  // X buffers dead -> reuse as tsims buffer

#pragma unroll
  for (int nt = 0; nt < 8; ++nt) {
    const int colt = nb + nt * 16;
    const int n = colt + lo;
    const int mbase = mt * 16 + 8 * hi;
    if (colt < B_SZ) {  // outputs: out[1 + n*C + c]
#pragma unroll
      for (int g = 0; g < 8; ++g)
        out[1 + (size_t)n * C_SZ + (c0 + mbase + g)] = accv[nt][g];
    } else {  // tsims -> LDS
      const int j = n - B_SZ;
#pragma unroll
      for (int g = 0; g < 8; ++g) tsbuf[(mbase + g) * 256 + j] = accv[nt][g];
    }
  }
  __syncthreads();

  // per-class mining: wave w owns local class rows 4w..4w+3
  float wsum = 0.f;
  for (int rr = 0; rr < 4; ++rr) {
    const int m = w * 4 + rr;
    const int c = c0 + m;
    float mn = BIGV;
#pragma unroll
    for (int it = 0; it < 8; ++it) {
      const int j = lane + 32 * it;
      const float v = tsbuf[m * 256 + j];
      if (targets[j] == c) mn = fminf(mn, v);
    }
#pragma unroll
    for (int off = 16; off > 0; off >>= 1) mn = fminf(mn, __shfl_xor(mn, off, 32));
    const bool has_pos = mn < 1e8f;
    const float thrd = has_pos ? (mn - MARGINF) : (1.0f - MARGINF);
    int cnt = 0;
    float s = 0.f;
#pragma unroll
    for (int it = 0; it < 8; ++it) {
      const int j = lane + 32 * it;
      const float v = tsbuf[m * 256 + j];
      const bool pos = (targets[j] == c);
      if (!pos && v > thrd) { cnt++; s += softplus_f(v); }
    }
#pragma unroll
    for (int off = 16; off > 0; off >>= 1) {
      cnt += __shfl_xor(cnt, off, 32);
      s += __shfl_xor(s, off, 32);
    }
    if (lane == 0) {
      const float thp_l = has_pos ? softplus_f(-mn) : 0.f;
      const float thn_l = (cnt > 0) ? s / (float)cnt : 0.f;
      wsum += thp_l + thn_l;
    }
  }
  if (lane == 0) atomicAdd(&acc[2], wsum * (1.0f / (float)C_SZ));
}

// ---------------------------------------------------------------------------
// Kernel 3: sim = nrm @ nrm.T (256x256, K=2048, f32 WMMA) + pair mining -> acc[1]
// ---------------------------------------------------------------------------
__global__ __launch_bounds__(256) void sim_mine_k(const float* __restrict__ X,
                                                  const unsigned char* __restrict__ posm,
                                                  const unsigned char* __restrict__ negm,
                                                  float* __restrict__ acc) {
  const int SKC = 32;
  __shared__ __align__(16) float As[16 * 32];
  __shared__ __align__(16) float Bs[256 * 32];
  __shared__ __align__(16) float simb[16 * 256];
  const int t = threadIdx.x, lane = t & 31, w = t >> 5;
  const int i0 = blockIdx.x * 16;
  const float* nrm = X + (size_t)B_SZ * D_SZ;
  const int hi = lane >> 4, lo = lane & 15;

  v8f accv[2];
  accv[0] = (v8f)0.f;
  accv[1] = (v8f)0.f;

#pragma unroll 1
  for (int k0 = 0; k0 < D_SZ; k0 += SKC) {
    __syncthreads();
    if (t < 128) {
      int r = t >> 3, kk = (t & 7) * 4;
      *(float4*)&As[r * SKC + kk] =
          *(const float4*)&nrm[(size_t)(i0 + r) * D_SZ + k0 + kk];
    }
#pragma unroll
    for (int i = 0; i < 8; ++i) {
      int p = t + 256 * i;
      int r = p >> 3, kk = (p & 7) * 4;
      *(float4*)&Bs[r * SKC + kk] =
          *(const float4*)&nrm[(size_t)r * D_SZ + k0 + kk];
    }
    __syncthreads();
#pragma unroll
    for (int ks = 0; ks < SKC; ks += 4) {
      v2f a = *(const v2f*)&As[lo * SKC + ks + 2 * hi];
#pragma unroll
      for (int nt = 0; nt < 2; ++nt) {
        v2f b = *(const v2f*)&Bs[(w * 32 + nt * 16 + lo) * SKC + ks + 2 * hi];
        accv[nt] = __builtin_amdgcn_wmma_f32_16x16x4_f32(
            false, a, false, b, (short)0, accv[nt], false, false);
      }
    }
  }
  __syncthreads();
#pragma unroll
  for (int nt = 0; nt < 2; ++nt) {
    const int n = w * 32 + nt * 16 + lo;
#pragma unroll
    for (int g = 0; g < 8; ++g) simb[(g + 8 * hi) * 256 + n] = accv[nt][g];
  }
  __syncthreads();

  float wsum = 0.f;
  for (int rr = 0; rr < 2; ++rr) {
    const int m = w * 2 + rr;
    const int ig = i0 + m;
    float mn = BIGV;
#pragma unroll
    for (int it = 0; it < 8; ++it) {
      const int j = lane + 32 * it;
      const float v = simb[m * 256 + j];
      if (posm[(size_t)ig * 256 + j]) mn = fminf(mn, v);
    }
#pragma unroll
    for (int off = 16; off > 0; off >>= 1) mn = fminf(mn, __shfl_xor(mn, off, 32));
    const float thrd = mn - MARGINF;
    int cnt = 0;
    float s = 0.f;
#pragma unroll
    for (int it = 0; it < 8; ++it) {
      const int j = lane + 32 * it;
      const float v = simb[m * 256 + j];
      if (negm[(size_t)ig * 256 + j] && v > thrd) { cnt++; s += softplus_f(v); }
    }
#pragma unroll
    for (int off = 16; off > 0; off >>= 1) {
      cnt += __shfl_xor(cnt, off, 32);
      s += __shfl_xor(s, off, 32);
    }
    if (lane == 0) {
      const float hp_l = softplus_f(-mn);
      const float hn_l = (cnt > 0) ? s / (float)cnt : 0.f;
      wsum += hp_l + hn_l;
    }
  }
  if (lane == 0) atomicAdd(&acc[1], wsum * (1.0f / (float)B_SZ));
}

// ---------------------------------------------------------------------------
// Kernel 4: log-softmax NLL over d_out rows -> acc[0]
// ---------------------------------------------------------------------------
__global__ __launch_bounds__(256) void bu_loss_k(const float* __restrict__ out,
                                                 const int* __restrict__ targets,
                                                 float* __restrict__ acc) {
  __shared__ float red[256];
  const int j = blockIdx.x, t = threadIdx.x;
  const float* row = out + 1 + (size_t)j * C_SZ;
  float mx = -INFINITY;
  for (int c = t; c < C_SZ; c += 256) mx = fmaxf(mx, row[c]);
  red[t] = mx;
  __syncthreads();
  for (int off = 128; off > 0; off >>= 1) {
    if (t < off) red[t] = fmaxf(red[t], red[t + off]);
    __syncthreads();
  }
  mx = red[0];
  __syncthreads();
  float se = 0.f;
  for (int c = t; c < C_SZ; c += 256) se += expf(row[c] - mx);
  red[t] = se;
  __syncthreads();
  for (int off = 128; off > 0; off >>= 1) {
    if (t < off) red[t] += red[t + off];
    __syncthreads();
  }
  if (t == 0) {
    const float lse = logf(red[0]) + mx;
    atomicAdd(&acc[0], (lse - row[targets[j]]) * (1.0f / (float)B_SZ));
  }
}

__global__ void finalize_k(const float* __restrict__ acc, float* __restrict__ out) {
  out[0] = acc[0] + acc[1] + acc[2];
}

// ---------------------------------------------------------------------------
extern "C" void kernel_launch(void* const* d_in, const int* in_sizes, int n_in,
                              void* d_out, int out_size, void* d_ws, size_t ws_size,
                              hipStream_t stream) {
  (void)in_sizes; (void)n_in; (void)out_size; (void)ws_size;
  const float* inputs = (const float*)d_in[0];
  const int* targets = (const int*)d_in[1];
  const unsigned char* posm = (const unsigned char*)d_in[2];
  const unsigned char* negm = (const unsigned char*)d_in[3];
  const float* V = (const float*)d_in[4];
  float* out = (float*)d_out;

  float* acc = (float*)d_ws;                       // acc[0]=bu,[1]=h,[2]=th
  float* Xf = acc + 64;                            // 512x2048 f32 (4 MB)
  bf16_t* Xh = (bf16_t*)(Xf + (size_t)512 * D_SZ); // 512x2048 bf16 (2 MB)
  bf16_t* Xl = Xh + (size_t)512 * D_SZ;            // 512x2048 bf16 (2 MB)

  prep_k<<<B_SZ, 256, 0, stream>>>(inputs, Xf, Xh, Xl, acc);
  vx_gemm_mine_k<<<C_SZ / 32, 256, SMEM_BYTES, stream>>>(V, Xh, Xl, targets, out, acc);
  sim_mine_k<<<B_SZ / 16, 256, 0, stream>>>(Xf, posm, negm, acc);
  bu_loss_k<<<B_SZ, 256, 0, stream>>>(out, targets, acc);
  finalize_k<<<1, 1, 0, stream>>>(acc, out);
}
